// MPNNConv_89163521065192
// MI455X (gfx1250) — matrix operand order repaired and checked by reading, not compiled, not run on hardware
//
#include <hip/hip_runtime.h>
#include <hip/hip_bf16.h>

typedef __attribute__((ext_vector_type(16))) __bf16 v16bf;
typedef __attribute__((ext_vector_type(8)))  float  v8f;

#define BN_EPS 1e-5f

__device__ __forceinline__ v8f wmma_bf16(v16bf a, v16bf b, v8f c) {
  // (neg_a, A, neg_b, B, c_mod, C, reuse_a, reuse_b)
  return __builtin_amdgcn_wmma_f32_16x16x32_bf16(false, a, false, b, (short)0, c,
                                                 false, false);
}

__device__ __forceinline__ void wait_async0() {
#if __has_builtin(__builtin_amdgcn_s_wait_asynccnt)
  __builtin_amdgcn_s_wait_asynccnt(0);
#else
  asm volatile("s_wait_asynccnt 0x0" ::: "memory");
#endif
}

// Async copy of 16B from global to LDS (gfx1250 GLOBAL_LOAD_ASYNC_TO_LDS_B128,
// tracked by ASYNCcnt; no VGPR data path).
__device__ __forceinline__ void async_b128(unsigned lds_byte, const float* gptr) {
  asm volatile("global_load_async_to_lds_b128 %0, %1, off"
               :: "v"(lds_byte), "v"((unsigned long long)(size_t)gptr)
               : "memory");
}

// B fragment: 32x16 tile of row-major f32 W[K][32], rows kbase..+31, cols nb..+15.
// B layout (table order): lanes 0-15 hold K=0..7,16..23 (elements 0..15),
// lanes 16-31 hold K=8..15,24..31; each lane owns one column.
__device__ __forceinline__ v16bf load_b_frag(const float* __restrict__ W,
                                             int kbase, int nb, int lane) {
  const int c   = nb + (lane & 15);
  const int khi = (lane & 16) ? 8 : 0;
  v16bf b;
#pragma unroll
  for (int i = 0; i < 16; ++i)
    b[i] = (__bf16)W[(unsigned)(kbase + khi + i + ((i >> 3) << 3)) * 32u + c];
  return b;
}

// ---------------------------------------------------------------------------
// Kernel 1: fused edge stage, double-buffered async-to-LDS pipeline.
//   msg  = relu([x_src | e] @ Wmsg + bmsg)            (K=64, 4 WMMAs / tile)
//   n_agg[dst] += msg                                  (f32 global atomics -> L2)
//   h1_e = (x_dst + x_src + (1+eps2)*e) @ eW1 + eb1    (K=32, 2 WMMAs / tile)
//   per-column sum / sumsq of h1_e for BatchNorm
// Each wave owns its LDS slice -> no __syncthreads; waves pipeline freely.
// ---------------------------------------------------------------------------
__global__ __launch_bounds__(256) void k_edge_fused(
    const float* __restrict__ x, const float* __restrict__ e,
    const int* __restrict__ src, const int* __restrict__ dst,
    const float* __restrict__ Wmsg, const float* __restrict__ bmsg,
    const float* __restrict__ eW1, const float* __restrict__ eb1,
    const float* __restrict__ eps2,
    float* __restrict__ n_agg, float* __restrict__ h1e,
    float* __restrict__ sum1e, float* __restrict__ sq1e, int tiles)
{
  __shared__ float S[2][8][3][16][32];  // [buf][wave][x_src,e,x_dst][row][col]
  const int lane = threadIdx.x & 31;
  const int wid  = threadIdx.x >> 5;
  const float c2 = 1.0f + eps2[0];

  // Issue the 48 row-copies (16 rows x {x_src, e, x_dst}) of one tile as
  // async global->LDS b128 transfers.
  auto issue_stage = [&](int buf, int tile) {
    const int base = tile * 16;
#pragma unroll
    for (int j = 0; j < 2; ++j) {
      const int job = lane + j * 32;
      if (job < 48) {
        const int kind = job >> 4, r = job & 15, row = base + r;
        const float* p = (kind == 0) ? x + (unsigned)src[row] * 32u
                       : (kind == 1) ? e + (unsigned)row * 32u
                                     : x + (unsigned)dst[row] * 32u;
        const unsigned lbase = (unsigned)(size_t)&S[buf][wid][kind][r][0];
#pragma unroll
        for (int q = 0; q < 8; ++q) async_b128(lbase + q * 16u, p + q * 4);
      }
    }
  };

  // Loop-invariant weight fragments + per-lane biases (registers).
  v16bf Bm[2][2], Be[2];
#pragma unroll
  for (int kc = 0; kc < 2; ++kc)
#pragma unroll
    for (int nh = 0; nh < 2; ++nh)
      Bm[kc][nh] = load_b_frag(Wmsg, kc * 32, nh * 16, lane);
#pragma unroll
  for (int nh = 0; nh < 2; ++nh)
    Be[nh] = load_b_frag(eW1, 0, nh * 16, lane);

  const int ar   = lane & 15;            // A row / C column-in-half
  const int khi  = (lane & 16) ? 8 : 0;  // A-fragment K offset
  const int roff = (lane & 16) ? 8 : 0;  // C-fragment row offset
  const int stride = gridDim.x * 8;

  float bmv[2], bev[2];                  // hoisted bias values (c fixed per lane)
#pragma unroll
  for (int nh = 0; nh < 2; ++nh) { bmv[nh] = bmsg[nh * 16 + ar]; bev[nh] = eb1[nh * 16 + ar]; }

  float s1[2] = {0.f, 0.f}, q1[2] = {0.f, 0.f};

  int buf = 0;
  {  // prologue: stage the first tile
    const int tile = blockIdx.x * 8 + wid;
    if (tile < tiles) issue_stage(0, tile);
  }
  for (int t0 = blockIdx.x * 8; t0 < tiles; t0 += stride) {
    const int tile = t0 + wid;
    if (tile < tiles) {                  // wave-uniform
      wait_async0();                     // current buffer resident in LDS
      const int ntile = t0 + stride + wid;
      if (ntile < tiles) issue_stage(buf ^ 1, ntile);  // prefetch next tile

      const int base = tile * 16;
      v16bf Ax, Aev, Ap;  // cat[0..31]=x_src, cat[32..63]=e, edge-MLP input
#pragma unroll
      for (int i = 0; i < 16; ++i) {
        const int k  = khi + i + ((i >> 3) << 3);
        const float xv = S[buf][wid][0][ar][k];
        const float ev = S[buf][wid][1][ar][k];
        const float dv = S[buf][wid][2][ar][k];
        Ax[i]  = (__bf16)xv;
        Aev[i] = (__bf16)ev;
        Ap[i]  = (__bf16)(dv + xv + c2 * ev);
      }
#pragma unroll
      for (int nh = 0; nh < 2; ++nh) {
        const int c = nh * 16 + ar;
        v8f accM, accE;
#pragma unroll
        for (int v = 0; v < 8; ++v) { accM[v] = bmv[nh]; accE[v] = bev[nh]; }
        accM = wmma_bf16(Ax,  Bm[0][nh], accM);   // K = 0..31  (x_src)
        accM = wmma_bf16(Aev, Bm[1][nh], accM);   // K = 32..63 (e)
        accE = wmma_bf16(Ap,  Be[nh],    accE);
#pragma unroll
        for (int v = 0; v < 8; ++v) {
          const int row = base + v + roff;
          // 32-bit element offsets -> saddr+voffset global forms
          atomicAdd(n_agg + ((unsigned)dst[row] * 32u + (unsigned)c),
                    fmaxf(accM[v], 0.f));
          const float hv = accE[v];
          h1e[(unsigned)row * 32u + (unsigned)c] = hv;
          s1[nh] += hv;  q1[nh] += hv * hv;
        }
      }
    }
    buf ^= 1;
  }
#pragma unroll
  for (int nh = 0; nh < 2; ++nh) {
    atomicAdd(sum1e + (nh * 16 + ar), s1[nh]);
    atomicAdd(sq1e  + (nh * 16 + ar), q1[nh]);
  }
}

// ---------------------------------------------------------------------------
// Kernel 2: generic [R,32] @ [32,32] + bias, with input preprocessing:
//   mode 0: a = A0 + (1+eps)*A1                       (node layer-1)
//   mode 1: a = relu(BN(A0)) using sum/sumsq stats    (layer-2, in-place safe)
// Accumulates per-column sum/sumsq of the result for the next BN.
// ---------------------------------------------------------------------------
__global__ __launch_bounds__(256) void k_lin32(
    const float* __restrict__ A0, const float* __restrict__ A1,
    const float* __restrict__ epsP,
    const float* __restrict__ ssum, const float* __restrict__ ssq, float invR,
    const float* __restrict__ gamma, const float* __restrict__ beta,
    const float* __restrict__ W, const float* __restrict__ bias,
    float* __restrict__ out, float* __restrict__ osum, float* __restrict__ osq,
    int tiles, int mode)
{
  __shared__ float S[8][16][32];
  __shared__ float scale[32], shift[32];
  const int lane = threadIdx.x & 31;
  const int wid  = threadIdx.x >> 5;

  if (threadIdx.x < 32 && mode == 1) {
    const float m   = ssum[threadIdx.x] * invR;
    const float var = ssq[threadIdx.x] * invR - m * m;
    const float sc  = rsqrtf(var + BN_EPS) * gamma[threadIdx.x];
    scale[threadIdx.x] = sc;
    shift[threadIdx.x] = beta[threadIdx.x] - m * sc;
  }
  const float cA = (mode == 0) ? (1.0f + epsP[0]) : 0.f;
  __syncthreads();

  v16bf B[2];
#pragma unroll
  for (int nh = 0; nh < 2; ++nh) B[nh] = load_b_frag(W, 0, nh * 16, lane);

  const int ar = lane & 15, khi = (lane & 16) ? 8 : 0, roff = (lane & 16) ? 8 : 0;
  float bv[2];                           // hoisted bias values
#pragma unroll
  for (int nh = 0; nh < 2; ++nh) bv[nh] = bias[nh * 16 + ar];
  float s[2] = {0.f, 0.f}, q[2] = {0.f, 0.f};

  for (int t0 = blockIdx.x * 8; t0 < tiles; t0 += gridDim.x * 8) {
    const int  tile   = t0 + wid;
    const bool active = tile < tiles;    // wave-uniform
    __syncthreads();
    if (active) {
      const unsigned base = (unsigned)tile * 512u;
#pragma unroll
      for (int j = 0; j < 16; ++j) {     // coalesced: lane = column
        const unsigned g = base + (unsigned)j * 32u + lane;
        S[wid][j][lane] = (mode == 0)
            ? A0[g] + cA * A1[g]
            : fmaxf(A0[g] * scale[lane] + shift[lane], 0.f);
      }
    }
    __syncthreads();
    if (active) {
      const unsigned rbase = (unsigned)tile * 16u;
      v16bf A;
#pragma unroll
      for (int i = 0; i < 16; ++i)
        A[i] = (__bf16)S[wid][ar][khi + i + ((i >> 3) << 3)];
#pragma unroll
      for (int nh = 0; nh < 2; ++nh) {
        const int c = nh * 16 + ar;
        v8f acc;
#pragma unroll
        for (int v = 0; v < 8; ++v) acc[v] = bv[nh];
        acc = wmma_bf16(A, B[nh], acc);
#pragma unroll
        for (int v = 0; v < 8; ++v) {
          const float hv = acc[v];
          out[(rbase + v + roff) * 32u + (unsigned)c] = hv;
          s[nh] += hv;  q[nh] += hv * hv;
        }
      }
    }
  }
#pragma unroll
  for (int nh = 0; nh < 2; ++nh) {
    atomicAdd(osum + (nh * 16 + ar), s[nh]);
    atomicAdd(osq  + (nh * 16 + ar), q[nh]);
  }
}

// ---------------------------------------------------------------------------
// Kernel 3: final BN + ReLU, elementwise, in-place safe.
// ---------------------------------------------------------------------------
__global__ __launch_bounds__(256) void k_bn_out(
    const float* __restrict__ h,
    const float* __restrict__ ssum, const float* __restrict__ ssq, float invR,
    const float* __restrict__ gamma, const float* __restrict__ beta,
    float* __restrict__ out, unsigned n)
{
  __shared__ float scale[32], shift[32];
  if (threadIdx.x < 32) {
    const float m   = ssum[threadIdx.x] * invR;
    const float var = ssq[threadIdx.x] * invR - m * m;
    const float sc  = rsqrtf(var + BN_EPS) * gamma[threadIdx.x];
    scale[threadIdx.x] = sc;
    shift[threadIdx.x] = beta[threadIdx.x] - m * sc;
  }
  __syncthreads();
  for (unsigned i = blockIdx.x * blockDim.x + threadIdx.x; i < n;
       i += gridDim.x * blockDim.x) {
    const int c = (int)(i & 31u);
    out[i] = fmaxf(h[i] * scale[c] + shift[c], 0.f);
  }
}

extern "C" void kernel_launch(void* const* d_in, const int* in_sizes, int n_in,
                              void* d_out, int out_size, void* d_ws, size_t ws_size,
                              hipStream_t stream) {
  const float* x    = (const float*)d_in[0];
  const float* e    = (const float*)d_in[1];
  const int*   ei   = (const int*)d_in[2];
  const float* Wmsg = (const float*)d_in[3];
  const float* bmsg = (const float*)d_in[4];
  const float* nW1  = (const float*)d_in[5];
  const float* nb1  = (const float*)d_in[6];
  const float* ng1  = (const float*)d_in[7];
  const float* nbe1 = (const float*)d_in[8];
  const float* nW2  = (const float*)d_in[9];
  const float* nb2  = (const float*)d_in[10];
  const float* ng2  = (const float*)d_in[11];
  const float* nbe2 = (const float*)d_in[12];
  const float* eW1  = (const float*)d_in[13];
  const float* eb1  = (const float*)d_in[14];
  const float* eg1  = (const float*)d_in[15];
  const float* ebe1 = (const float*)d_in[16];
  const float* eW2  = (const float*)d_in[17];
  const float* eb2  = (const float*)d_in[18];
  const float* eg2  = (const float*)d_in[19];
  const float* ebe2 = (const float*)d_in[20];
  const float* eps1 = (const float*)d_in[21];
  const float* eps2 = (const float*)d_in[22];

  const int N = in_sizes[0] / 32;       // 50000 (divisible by 16)
  const int E = in_sizes[1] / 32;       // 1600000 (divisible by 16)
  const int* src = ei;                  // edge_index[0]
  const int* dst = ei + E;              // edge_index[1]

  // Workspace: n_agg [N*32] + 8 x 32-float BN stat accumulators.
  float* n_agg = (float*)d_ws;
  float* st    = n_agg + (size_t)N * 32;
  float *sum1e = st + 0,   *sq1e = st + 32,  *sum2e = st + 64,  *sq2e = st + 96;
  float *sum1n = st + 128, *sq1n = st + 160, *sum2n = st + 192, *sq2n = st + 224;

  // d_out = [n_out (N*32) | e_out (E*32)]; both regions double as activation
  // scratch (h1 -> h2 -> final, each update row-wise in place).
  float* n_reg = (float*)d_out;
  float* e_reg = n_reg + (size_t)N * 32;

  hipMemsetAsync(d_ws, 0, ((size_t)N * 32 + 256) * sizeof(float), stream);

  const int tilesE = E / 16, tilesN = N / 16;
  int gE = (tilesE + 7) / 8; if (gE > 2048) gE = 2048;
  int gN = (tilesN + 7) / 8; if (gN > 2048) gN = 2048;

  // 1) message + scatter + edge layer-1 (+ its BN stats)
  k_edge_fused<<<gE, 256, 0, stream>>>(x, e, src, dst, Wmsg, bmsg, eW1, eb1,
                                       eps2, n_agg, e_reg, sum1e, sq1e, tilesE);
  // 2) node layer-1: (n_agg + (1+eps1)*x) @ nW1 + nb1
  k_lin32<<<gN, 256, 0, stream>>>(n_agg, x, eps1, nullptr, nullptr, 0.f,
                                  nullptr, nullptr, nW1, nb1,
                                  n_reg, sum1n, sq1n, tilesN, 0);
  // 3) edge layer-2: relu(BN(h1_e)) @ eW2 + eb2   (in place)
  k_lin32<<<gE, 256, 0, stream>>>(e_reg, nullptr, nullptr, sum1e, sq1e,
                                  1.0f / (float)E, eg1, ebe1, eW2, eb2,
                                  e_reg, sum2e, sq2e, tilesE, 1);
  // 4) node layer-2
  k_lin32<<<gN, 256, 0, stream>>>(n_reg, nullptr, nullptr, sum1n, sq1n,
                                  1.0f / (float)N, ng1, nbe1, nW2, nb2,
                                  n_reg, sum2n, sq2n, tilesN, 1);
  // 5) final BN+ReLU in place
  k_bn_out<<<2048, 256, 0, stream>>>(e_reg, sum2e, sq2e, 1.0f / (float)E,
                                     eg2, ebe2, e_reg, (unsigned)E * 32u);
  k_bn_out<<<512, 256, 0, stream>>>(n_reg, sum2n, sq2n, 1.0f / (float)N,
                                    ng2, nbe2, n_reg, (unsigned)N * 32u);
}
